// MambaFull_75187697483953
// MI455X (gfx1250) — compile-verified, matrix operand order checked by reading, not run on hardware
//
#include <hip/hip_runtime.h>
#include <hip/hip_bf16.h>
#include <math.h>

// ---------------------------------------------------------------------------
// MI455X / gfx1250 Mamba forward.
// GEMMs: bf16 operands, f32 accumulation via v_wmma_f32_16x16x32_bf16.
// Scan / norms: f32 VALU. All launches on `stream`, scratch from d_ws only.
// ---------------------------------------------------------------------------

typedef __bf16 bf16_t;
typedef bf16_t v16bf __attribute__((ext_vector_type(16)));
typedef bf16_t v8bf  __attribute__((ext_vector_type(8)));
typedef float  v8f   __attribute__((ext_vector_type(8)));

// Problem constants (match setup_inputs()).
#define BSZ   32
#define LSEQ  1024
#define DMODEL 256
#define DINNER 512
#define DSTATE 16
#define DCONV  4
#define DTRANK 16
#define NBLK   3
#define NCLS   1024
#define NROWS  (BSZ * LSEQ)          // 32768 flattened tokens
#define XPROJ  (DTRANK + 2 * DSTATE) // 48

// ---- helpers --------------------------------------------------------------

__device__ __forceinline__ bf16_t f2bf(float f) {
  unsigned u = __builtin_bit_cast(unsigned, f);
  unsigned r = (u + 0x7FFFu + ((u >> 16) & 1u)) >> 16;   // RNE
  return __builtin_bit_cast(bf16_t, (unsigned short)r);
}

// Load a 16(M/N) x 32(K) bf16 fragment from a row-major matrix, per the
// CDNA5 16-bit A/B VGPR layout (ISA 7.12.2). Branch-free; two b128 loads.
__device__ __forceinline__ v16bf load_frag16x32(const bf16_t* __restrict__ p, int ld) {
  int lane = threadIdx.x & 31;
  int row  = lane & 15;
  int kb   = (lane >> 4) * 8;                 // lanes 16..31 hold K+8 group
  const bf16_t* q = p + (size_t)row * ld + kb;
  v8bf lo = *(const v8bf*)(q);                // K = kb .. kb+7
  v8bf hi = *(const v8bf*)(q + 16);           // K = kb+16 .. kb+23
  v16bf f;
#pragma unroll
  for (int i = 0; i < 8; ++i) { f[i] = lo[i]; f[8 + i] = hi[i]; }
  return f;
}

// ---- generic bf16 WMMA GEMM: C(MxN,f32) = A(MxK) * W(NxK)^T ---------------
// Each wave owns one 16-row M tile and NT consecutive 16-col N tiles,
// reusing the A fragment NT times per K step.
template <int NT>
__global__ __launch_bounds__(256)
void wmma_gemm_bf16(const bf16_t* __restrict__ A, const bf16_t* __restrict__ W,
                    float* __restrict__ C, int M, int N, int K,
                    int lda, int ldw, int ldc) {
  int wave = (int)(blockIdx.x * (blockDim.x >> 5) + (threadIdx.x >> 5));
  int ngrp = N / (16 * NT);
  int mt   = wave / ngrp;
  int ng   = wave % ngrp;
  if (mt * 16 >= M) return;                   // wave-uniform: EXEC stays full

  v8f acc[NT];
#pragma unroll
  for (int t = 0; t < NT; ++t) acc[t] = v8f{0.f,0.f,0.f,0.f,0.f,0.f,0.f,0.f};

  const bf16_t* Ap = A + (size_t)(mt * 16) * lda;
  for (int k = 0; k < K; k += 32) {
    __builtin_prefetch(Ap + k + 64, 0, 1);    // global_prefetch_b8 -> L2 ahead
    v16bf af = load_frag16x32(Ap + k, lda);
#pragma unroll
    for (int t = 0; t < NT; ++t) {
      const bf16_t* Wp = W + (size_t)((ng * NT + t) * 16) * ldw + k;
      v16bf bf = load_frag16x32(Wp, ldw);
      acc[t] = __builtin_amdgcn_wmma_f32_16x16x32_bf16(
          /*neg_a=*/false, af, /*neg_b=*/false, bf,
          /*c_mod=*/(short)0, acc[t], /*reuse_a=*/false, /*reuse_b=*/false);
    }
  }

  int lane = threadIdx.x & 31;
  int n0   = lane & 15;
  int mb   = (lane >> 4) * 8;                 // VGPR r: M=r (lo lanes) / r+8
#pragma unroll
  for (int t = 0; t < NT; ++t) {
    float* Cp = C + (size_t)(mt * 16 + mb) * ldc + (ng * NT + t) * 16 + n0;
#pragma unroll
    for (int r = 0; r < 8; ++r) Cp[(size_t)r * ldc] = acc[t][r];
  }
}

// ---- elementwise / prep kernels ------------------------------------------

__global__ void cvt_f32_bf16(const float* __restrict__ in, bf16_t* __restrict__ out, int n) {
  int i = blockIdx.x * blockDim.x + threadIdx.x;
  if (i < n) out[i] = f2bf(in[i]);
}

// dt_w (nb, DINNER, DTRANK) -> bf16 (nb, DINNER, 32) zero-padded in K so we
// can reuse the K=32 WMMA path while A (x_dbl, ld=48) carries junk in K>=16.
__global__ void cvt_dtw_pad(const float* __restrict__ in, bf16_t* __restrict__ out) {
  int i = blockIdx.x * blockDim.x + threadIdx.x;     // over NBLK*DINNER*32
  if (i >= NBLK * DINNER * 32) return;
  int k = i & 31, row = i >> 5;
  out[i] = (k < DTRANK) ? f2bf(in[row * DTRANK + k]) : f2bf(0.f);
}

__global__ void embed_kernel(const float* __restrict__ x, const float* __restrict__ We,
                             const float* __restrict__ be, float* __restrict__ h) {
  int i = blockIdx.x * blockDim.x + threadIdx.x;     // over NROWS*DMODEL
  if (i >= NROWS * DMODEL) return;
  int d = i & (DMODEL - 1), bl = i >> 8;
  h[i] = x[bl * 2 + 0] * We[d * 2 + 0] + x[bl * 2 + 1] * We[d * 2 + 1] + be[d];
}

// residual (+)= h ; hn = LayerNorm(residual)*w + b  -> bf16 A operand.
__global__ __launch_bounds__(DMODEL)
void ln_fused(const float* __restrict__ h, float* __restrict__ residual,
              const float* __restrict__ w, const float* __restrict__ b,
              bf16_t* __restrict__ hn, int first) {
  int row = blockIdx.x, c = threadIdx.x;
  int idx = row * DMODEL + c;
  float r = first ? h[idx] : (residual[idx] + h[idx]);
  residual[idx] = r;
  __shared__ float red[DMODEL];
  red[c] = r; __syncthreads();
  for (int s = DMODEL / 2; s > 0; s >>= 1) { if (c < s) red[c] += red[c + s]; __syncthreads(); }
  float mu = red[0] * (1.f / DMODEL);
  __syncthreads();
  float d0 = r - mu;
  red[c] = d0 * d0; __syncthreads();
  for (int s = DMODEL / 2; s > 0; s >>= 1) { if (c < s) red[c] += red[c + s]; __syncthreads(); }
  float var = red[0] * (1.f / DMODEL);
  hn[idx] = f2bf(d0 * rsqrtf(var + 1e-5f) * w[c] + b[c]);
}

// depthwise causal conv (dc=4) over x = xz[:, :DINNER] + SiLU. Emits f32 (scan)
// and bf16 (x_proj A operand).
__global__ void conv_silu(const float* __restrict__ xz, const float* __restrict__ cw,
                          const float* __restrict__ cb, float* __restrict__ u,
                          bf16_t* __restrict__ ubf) {
  int i = blockIdx.x * blockDim.x + threadIdx.x;     // over NROWS*DINNER
  if (i >= NROWS * DINNER) return;
  int e = i & (DINNER - 1), bl = i >> 9;
  int l = bl & (LSEQ - 1);
  float acc = cb[e];
#pragma unroll
  for (int j = 0; j < DCONV; ++j) {
    int ll = l - (DCONV - 1) + j;
    if (ll >= 0) acc += xz[(size_t)(bl - (DCONV - 1) + j) * (2 * DINNER) + e] * cw[e * DCONV + j];
  }
  float s = acc / (1.f + __expf(-acc));
  u[i] = s; ubf[i] = f2bf(s);
}

__global__ void dt_softplus(float* __restrict__ dtraw, const float* __restrict__ dtb) {
  int i = blockIdx.x * blockDim.x + threadIdx.x;     // over NROWS*DINNER
  if (i >= NROWS * DINNER) return;
  float v = dtraw[i] + dtb[i & (DINNER - 1)];
  dtraw[i] = (v > 20.f) ? v : log1pf(__expf(v));
}

// Selective scan: sequential in L, parallel over (b, d). 16 f32 states per
// thread in registers; B/C broadcast through LDS. Fuses +D*u and SiLU(z) gate,
// writes bf16 y (out_proj A operand).
__global__ __launch_bounds__(256)
void scan_kernel(const float* __restrict__ dt, const float* __restrict__ u,
                 const float* __restrict__ xdbl, const float* __restrict__ xz,
                 const float* __restrict__ A_log, const float* __restrict__ Dp,
                 bf16_t* __restrict__ ybf) {
  int b = blockIdx.y;
  int d = blockIdx.x * blockDim.x + threadIdx.x;     // 0..DINNER-1
  float Arow[DSTATE];
#pragma unroll
  for (int s = 0; s < DSTATE; ++s) Arow[s] = -__expf(A_log[d * DSTATE + s]);
  float Dd = Dp[d];
  float hst[DSTATE];
#pragma unroll
  for (int s = 0; s < DSTATE; ++s) hst[s] = 0.f;
  __shared__ float sB[DSTATE], sC[DSTATE];

  for (int t = 0; t < LSEQ; ++t) {
    size_t row = (size_t)b * LSEQ + t;
    if (threadIdx.x < 32) {
      int s = threadIdx.x & 15;
      float v = xdbl[row * XPROJ + DTRANK + ((threadIdx.x >> 4) * DSTATE) + s];
      if (threadIdx.x < 16) sB[s] = v; else sC[s] = v;
    }
    __syncthreads();
    float dtv = dt[row * DINNER + d];
    float uv  = u[row * DINNER + d];
    float dtu = dtv * uv;
    float acc = 0.f;
#pragma unroll
    for (int s = 0; s < DSTATE; ++s) {
      float dA = __expf(dtv * Arow[s]);
      hst[s] = dA * hst[s] + dtu * sB[s];
      acc += hst[s] * sC[s];
    }
    float y = acc + Dd * uv;
    float z = xz[row * (2 * DINNER) + DINNER + d];
    float g = z / (1.f + __expf(-z));
    ybf[row * DINNER + d] = f2bf(y * g);
    __syncthreads();
  }
}

// residual += h ; RMSNorm * w + b -> bf16 head-GEMM operand.
__global__ __launch_bounds__(DMODEL)
void rms_fused(const float* __restrict__ h, float* __restrict__ residual,
               const float* __restrict__ w, const float* __restrict__ b,
               bf16_t* __restrict__ normed) {
  int row = blockIdx.x, c = threadIdx.x;
  int idx = row * DMODEL + c;
  float r = residual[idx] + h[idx];
  residual[idx] = r;
  __shared__ float red[DMODEL];
  red[c] = r * r; __syncthreads();
  for (int s = DMODEL / 2; s > 0; s >>= 1) { if (c < s) red[c] += red[c + s]; __syncthreads(); }
  float rms = rsqrtf(red[0] * (1.f / DMODEL) + 1e-5f);
  normed[idx] = f2bf(r * rms * w[c] + b[c]);
}

// ---- host side ------------------------------------------------------------

static inline int cdiv(long long a, long long b) { return (int)((a + b - 1) / b); }

template <int NT>
static void launch_gemm(const bf16_t* A, const bf16_t* W, float* C,
                        int M, int N, int K, int lda, int ldw, int ldc,
                        hipStream_t stream) {
  long long waves = (long long)(M / 16) * (N / (16 * NT));
  int blocks = cdiv(waves * 32, 256);
  wmma_gemm_bf16<NT><<<blocks, 256, 0, stream>>>(A, W, C, M, N, K, lda, ldw, ldc);
}

extern "C" void kernel_launch(void* const* d_in, const int* in_sizes, int n_in,
                              void* d_out, int out_size, void* d_ws, size_t ws_size,
                              hipStream_t stream) {
  const float* x      = (const float*)d_in[0];
  const float* We     = (const float*)d_in[1];
  const float* be     = (const float*)d_in[2];
  const float* ln_w   = (const float*)d_in[3];
  const float* ln_b   = (const float*)d_in[4];
  const float* in_w   = (const float*)d_in[5];
  const float* conv_w = (const float*)d_in[6];
  const float* conv_b = (const float*)d_in[7];
  const float* xp_w   = (const float*)d_in[8];
  const float* dt_w   = (const float*)d_in[9];
  const float* dt_b   = (const float*)d_in[10];
  const float* A_log  = (const float*)d_in[11];
  const float* Dmat   = (const float*)d_in[12];
  const float* out_w  = (const float*)d_in[13];
  const float* normfw = (const float*)d_in[14];
  const float* normfb = (const float*)d_in[15];
  const float* head_w = (const float*)d_in[16];
  float* out = (float*)d_out;

  // bump allocator over d_ws
  char* ws = (char*)d_ws;
  size_t off = 0;
  auto alloc = [&](size_t bytes) -> void* {
    void* p = ws + off;
    off = (off + bytes + 255) & ~(size_t)255;
    return p;
  };

  float*  residual = (float*)alloc((size_t)NROWS * DMODEL * 4);
  float*  hbuf     = (float*)alloc((size_t)NROWS * DMODEL * 4);
  bf16_t* hn_bf    = (bf16_t*)alloc((size_t)NROWS * DMODEL * 2);   // also reused for final normed
  float*  xz       = (float*)alloc((size_t)NROWS * 2 * DINNER * 4);
  float*  ubuf     = (float*)alloc((size_t)NROWS * DINNER * 4);
  bf16_t* u_bf     = (bf16_t*)alloc((size_t)NROWS * DINNER * 2);
  float*  xdbl     = (float*)alloc((size_t)NROWS * XPROJ * 4);
  bf16_t* xdbl_bf  = (bf16_t*)alloc((size_t)NROWS * XPROJ * 2);
  float*  dtbuf    = (float*)alloc((size_t)NROWS * DINNER * 4);
  bf16_t* y_bf     = (bf16_t*)alloc((size_t)NROWS * DINNER * 2);
  bf16_t* inw_bf   = (bf16_t*)alloc((size_t)NBLK * 2 * DINNER * DMODEL * 2);
  bf16_t* xpw_bf   = (bf16_t*)alloc((size_t)NBLK * XPROJ * DINNER * 2);
  bf16_t* dtw_bf   = (bf16_t*)alloc((size_t)NBLK * DINNER * 32 * 2);
  bf16_t* outw_bf  = (bf16_t*)alloc((size_t)NBLK * DMODEL * DINNER * 2);
  bf16_t* headw_bf = (bf16_t*)alloc((size_t)NCLS * DMODEL * 2);

  // ---- weight precision conversion (runs every call; deterministic) ----
  {
    int n;
    n = NBLK * 2 * DINNER * DMODEL;
    cvt_f32_bf16<<<cdiv(n, 256), 256, 0, stream>>>(in_w, inw_bf, n);
    n = NBLK * XPROJ * DINNER;
    cvt_f32_bf16<<<cdiv(n, 256), 256, 0, stream>>>(xp_w, xpw_bf, n);
    n = NBLK * DMODEL * DINNER;
    cvt_f32_bf16<<<cdiv(n, 256), 256, 0, stream>>>(out_w, outw_bf, n);
    n = NCLS * DMODEL;
    cvt_f32_bf16<<<cdiv(n, 256), 256, 0, stream>>>(head_w, headw_bf, n);
    n = NBLK * DINNER * 32;
    cvt_dtw_pad<<<cdiv(n, 256), 256, 0, stream>>>(dt_w, dtw_bf);
  }

  // ---- embedding ----
  embed_kernel<<<cdiv((long long)NROWS * DMODEL, 256), 256, 0, stream>>>(x, We, be, hbuf);

  // ---- mamba blocks ----
  for (int i = 0; i < NBLK; ++i) {
    // residual + LayerNorm -> bf16
    ln_fused<<<NROWS, DMODEL, 0, stream>>>(hbuf, residual,
                                           ln_w + i * DMODEL, ln_b + i * DMODEL,
                                           hn_bf, i == 0 ? 1 : 0);
    // in_proj: (32768 x 256) x (1024 x 256)^T -> xz
    launch_gemm<4>(hn_bf, inw_bf + (size_t)i * 2 * DINNER * DMODEL, xz,
                   NROWS, 2 * DINNER, DMODEL, DMODEL, DMODEL, 2 * DINNER, stream);
    // depthwise conv + SiLU
    conv_silu<<<cdiv((long long)NROWS * DINNER, 256), 256, 0, stream>>>(
        xz, conv_w + (size_t)i * DINNER * DCONV, conv_b + (size_t)i * DINNER, ubuf, u_bf);
    // x_proj: (32768 x 512) x (48 x 512)^T -> x_dbl
    launch_gemm<1>(u_bf, xpw_bf + (size_t)i * XPROJ * DINNER, xdbl,
                   NROWS, XPROJ, DINNER, DINNER, DINNER, XPROJ, stream);
    {
      int n = NROWS * XPROJ;
      cvt_f32_bf16<<<cdiv(n, 256), 256, 0, stream>>>(xdbl, xdbl_bf, n);
    }
    // dt_proj: (32768 x 48[bf16, K window 0..31]) x (512 x 32)^T -> dt_raw
    launch_gemm<4>(xdbl_bf, dtw_bf + (size_t)i * DINNER * 32, dtbuf,
                   NROWS, DINNER, 32, XPROJ, 32, DINNER, stream);
    dt_softplus<<<cdiv((long long)NROWS * DINNER, 256), 256, 0, stream>>>(
        dtbuf, dt_b + (size_t)i * DINNER);
    // selective scan (+D*u, *SiLU(z)) -> y bf16
    {
      dim3 grid(DINNER / 256, BSZ);
      scan_kernel<<<grid, 256, 0, stream>>>(dtbuf, ubuf, xdbl, xz,
                                            A_log + (size_t)i * DINNER * DSTATE,
                                            Dmat + (size_t)i * DINNER, y_bf);
    }
    // out_proj: (32768 x 512) x (256 x 512)^T -> h
    launch_gemm<4>(y_bf, outw_bf + (size_t)i * DMODEL * DINNER, hbuf,
                   NROWS, DMODEL, DINNER, DINNER, DINNER, DMODEL, stream);
  }

  // ---- final residual + RMSNorm -> bf16 ; head GEMM -> d_out (f32) ----
  rms_fused<<<NROWS, DMODEL, 0, stream>>>(hbuf, residual, normfw, normfb, hn_bf);
  launch_gemm<4>(hn_bf, headw_bf, out, NROWS, NCLS, DMODEL, DMODEL, DMODEL, NCLS, stream);
}
